// BinaryFullTensorCell_75453985456646
// MI455X (gfx1250) — matrix operand order repaired and checked by reading, not compiled, not run on hardware
//
#include <hip/hip_runtime.h>
#include <hip/hip_bf16.h>

// ---------------------------------------------------------------------------
// BinaryFullTensorCell on MI455X (gfx1250)
//   out = (h1 (x) h2) @ A  + h1 @ U1^T + h2 @ U2^T + b
// GEMM view: M=16384, K=16384 (i*128+j), N=384, left rows rank-1 generated.
// Compute-bound (206 GFLOP vs 67MB traffic) -> bf16 WMMA 16x16x32, f32 accum.
// ---------------------------------------------------------------------------

typedef __attribute__((ext_vector_type(16))) __bf16 v16bf;
typedef __attribute__((ext_vector_type(8)))  float  v8f;

#define HD        128          // hidden dim
#define NNODES    16384
#define NOUT      384          // 3*HD
#define BM        128          // rows per workgroup tile
#define LDH       132          // padded LDS row stride (floats): 528B, 16B aligned, bank-safe
#define NB_TOTAL  24           // 384/16 column blocks
#define NFRAG_A   (HD * 4 * NB_TOTAL)          // 12288 product fragments
#define NFRAG_U   (4 * NB_TOTAL)               // 96 per U matrix
#define NFRAG_TOT (NFRAG_A + 2 * NFRAG_U)      // 12480
// fragment = 32 lanes x 16 bf16 = 1024 bytes = 512 u16

// ---------------------------------------------------------------------------
// Pre-pass: swizzle A / U1^T / U2^T into bf16 WMMA B-fragments (32x16).
// B layout used here (contiguous k, not AMD's table order): lane l holds column n = l%16;
//   lanes 0-15: VGPR v holds K=2v,2v+1 (K 0..15); lanes 16-31: K 16..31.
// ---------------------------------------------------------------------------
__global__ void btc_prep_kernel(const float* __restrict__ A,
                                const float* __restrict__ U1,
                                const float* __restrict__ U2,
                                unsigned short* __restrict__ ws) {
    const int wave = threadIdx.x >> 5;
    const int lane = threadIdx.x & 31;
    const int f    = blockIdx.x * 8 + wave;
    if (f >= NFRAG_TOT) return;

    const int kh   = (lane >> 4) & 1;   // K-half: 0 -> K 0..15, 1 -> K 16..31
    const int ncol = lane & 15;

    v16bf frag;
    if (f < NFRAG_A) {
        // product fragment: (i, t, nb); rows j = t*32 + kh*16 + e, col o
        const int i  = f / (4 * NB_TOTAL);
        const int r  = f % (4 * NB_TOTAL);
        const int t  = r / NB_TOTAL;
        const int nb = r % NB_TOTAL;
        const int col   = nb * 16 + ncol;
        const int jbase = t * 32 + kh * 16;
        const float* src = A + (size_t)(i * HD + jbase) * NOUT + col;
#pragma unroll
        for (int e = 0; e < 16; ++e)
            frag[e] = (__bf16)src[(size_t)e * NOUT];
    } else {
        // U fragment: B[d][o] = U[o*HD + d]   (U @ .T -> transpose on gather)
        const int g  = f - NFRAG_A;
        const float* U = (g < NFRAG_U) ? U1 : U2;
        const int r  = g % NFRAG_U;
        const int t  = r / NB_TOTAL;
        const int nb = r % NB_TOTAL;
        const int col   = nb * 16 + ncol;
        const int dbase = t * 32 + kh * 16;
        const float* src = U + (size_t)col * HD + dbase;
#pragma unroll
        for (int e = 0; e < 16; ++e)
            frag[e] = (__bf16)src[e];
    }
    *reinterpret_cast<v16bf*>(ws + (size_t)f * 512 + lane * 16) = frag;
}

// ---------------------------------------------------------------------------
// A-fragment build (contiguous k order matching the B fragments, 16x32):
//   lanes 0-15:  M=lane,   V0..7 = K 0..15
//   lanes 16-31: M=lane-16,V0..7 = K 16..31
// value = s * hrow[jbase + K]   (product rounded once to bf16)
// ---------------------------------------------------------------------------
__device__ __forceinline__ v16bf make_afrag(const float* __restrict__ hrow,
                                            float s, int jbase, int kh) {
    v16bf a;
#pragma unroll
    for (int v = 0; v < 8; ++v) {
        const int kl = 2 * v + kh * 16;
        const float2 p = *reinterpret_cast<const float2*>(hrow + jbase + kl);
        a[2 * v]     = (__bf16)(s * p.x);
        a[2 * v + 1] = (__bf16)(s * p.y);
    }
    return a;
}

// ---------------------------------------------------------------------------
// Main kernel: 256 threads = 8 waves (4M x 2N), WG tile 128x128.
// ---------------------------------------------------------------------------
__global__ void __launch_bounds__(256, 1)
btc_main_kernel(const float* __restrict__ nh,
                const unsigned short* __restrict__ ws,
                const float* __restrict__ bias,
                float* __restrict__ out) {
    extern __shared__ float smem[];
    float* lh1 = smem;              // [BM][LDH]
    float* lh2 = smem + BM * LDH;   // [BM][LDH]

    const int tid   = threadIdx.x;
    const int lane  = tid & 31;
    const int wave  = tid >> 5;
    const int wm    = wave & 3;     // wave M index (32 rows each)
    const int wn    = wave >> 2;    // wave N index (64 cols each)
    const int mtile = blockIdx.x;   // 0..127
    const int cb    = blockIdx.y;   // 0..2 (128-col block)
    const int m0    = mtile * BM;

    // ---- stage h1/h2 tiles (fp32) into LDS: 128 rows x 32 float4 each ----
    for (int idx = tid; idx < BM * (HD / 4); idx += 256) {
        const int row = idx >> 5;
        const int c4  = idx & 31;
        const float4 v1 = *(reinterpret_cast<const float4*>(
                              nh + ((size_t)(m0 + row) * 2 + 0) * HD) + c4);
        const float4 v2 = *(reinterpret_cast<const float4*>(
                              nh + ((size_t)(m0 + row) * 2 + 1) * HD) + c4);
        *reinterpret_cast<float4*>(lh1 + row * LDH + c4 * 4) = v1;
        *reinterpret_cast<float4*>(lh2 + row * LDH + c4 * 4) = v2;
    }
    __syncthreads();

    v8f acc[2][4] = {};             // 2 M-subtiles x 4 N-subtiles, f32 accum

    const int kh   = (lane >> 4) & 1;
    const int mloc = lane & 15;
    const int nbg0 = cb * 8 + wn * 4;            // wave's global 16-col block
    const float* h2r0 = lh2 + (wm * 32 + mloc) * LDH;
    const float* h2r1 = lh2 + (wm * 32 + 16 + mloc) * LDH;
    const float* h1r0 = lh1 + (wm * 32 + mloc) * LDH;
    const float* h1r1 = lh1 + (wm * 32 + 16 + mloc) * LDH;

    // ---- main K loop: 128 chunks, chunk i covers j=0..127 (K=128 = 4 steps) ----
    for (int i = 0; i < HD; ++i) {
        const float s0 = h1r0[i];                // h1[row, i] per A-frag row
        const float s1 = h1r1[i];
#pragma unroll
        for (int t = 0; t < 4; ++t) {
            v16bf b[4];
#pragma unroll
            for (int nb = 0; nb < 4; ++nb) {
                const size_t f = (size_t)(i * 4 + t) * NB_TOTAL + (nbg0 + nb);
                b[nb] = *reinterpret_cast<const v16bf*>(ws + f * 512 + lane * 16);
            }
            const v16bf a0 = make_afrag(h2r0, s0, t * 32, kh);
            const v16bf a1 = make_afrag(h2r1, s1, t * 32, kh);
#pragma unroll
            for (int nb = 0; nb < 4; ++nb) {
                acc[0][nb] = __builtin_amdgcn_wmma_f32_16x16x32_bf16(
                    false, a0, false, b[nb], (short)0, acc[0][nb], false, false);
                acc[1][nb] = __builtin_amdgcn_wmma_f32_16x16x32_bf16(
                    false, a1, false, b[nb], (short)0, acc[1][nb], false, false);
            }
        }
    }

    // ---- linear terms as two extra K-chunks: u=0 -> h1@U1^T, u=1 -> h2@U2^T ----
#pragma unroll
    for (int u = 0; u < 2; ++u) {
        const float* hr0 = (u == 0) ? h1r0 : h2r0;
        const float* hr1 = (u == 0) ? h1r1 : h2r1;
#pragma unroll
        for (int t = 0; t < 4; ++t) {
            v16bf b[4];
#pragma unroll
            for (int nb = 0; nb < 4; ++nb) {
                const size_t f = (size_t)(NFRAG_A + u * NFRAG_U)
                               + (size_t)t * NB_TOTAL + (nbg0 + nb);
                b[nb] = *reinterpret_cast<const v16bf*>(ws + f * 512 + lane * 16);
            }
            const v16bf a0 = make_afrag(hr0, 1.0f, t * 32, kh);
            const v16bf a1 = make_afrag(hr1, 1.0f, t * 32, kh);
#pragma unroll
            for (int nb = 0; nb < 4; ++nb) {
                acc[0][nb] = __builtin_amdgcn_wmma_f32_16x16x32_bf16(
                    false, a0, false, b[nb], (short)0, acc[0][nb], false, false);
                acc[1][nb] = __builtin_amdgcn_wmma_f32_16x16x32_bf16(
                    false, a1, false, b[nb], (short)0, acc[1][nb], false, false);
            }
        }
    }

    // ---- epilogue: + bias, store fp32.  C layout: lane n fixed = base+lane%16;
    //      VGPR r -> M = r (lanes 0-15) / 8+r (lanes 16-31). ----
    const int mrow0 = m0 + wm * 32;
    const int moff  = kh * 8;
#pragma unroll
    for (int nb = 0; nb < 4; ++nb) {
        const int n  = (nbg0 + nb) * 16 + mloc;
        const float bv = bias[n];
#pragma unroll
        for (int mb = 0; mb < 2; ++mb) {
            const int mr = mrow0 + mb * 16 + moff;
#pragma unroll
            for (int r = 0; r < 8; ++r)
                out[(size_t)(mr + r) * NOUT + n] = acc[mb][nb][r] + bv;
        }
    }
}

// ---------------------------------------------------------------------------
// Fallback (no workspace): naive fp32, correct but slow. Only if ws too small.
// ---------------------------------------------------------------------------
__global__ void btc_fallback_kernel(const float* __restrict__ nh,
                                    const float* __restrict__ A,
                                    const float* __restrict__ U1,
                                    const float* __restrict__ U2,
                                    const float* __restrict__ bias,
                                    float* __restrict__ out) {
    const int m = blockIdx.x;
    const float* h1 = nh + (size_t)m * 2 * HD;
    const float* h2 = h1 + HD;
    for (int o = threadIdx.x; o < NOUT; o += blockDim.x) {
        float acc = bias[o];
        for (int i = 0; i < HD; ++i) {
            acc += h1[i] * U1[(size_t)o * HD + i] + h2[i] * U2[(size_t)o * HD + i];
            const float* Ai = A + (size_t)i * HD * NOUT + o;
            float t = 0.f;
            for (int j = 0; j < HD; ++j) t += h2[j] * Ai[(size_t)j * NOUT];
            acc += h1[i] * t;
        }
        out[(size_t)m * NOUT + o] = acc;
    }
}

// ---------------------------------------------------------------------------
extern "C" void kernel_launch(void* const* d_in, const int* in_sizes, int n_in,
                              void* d_out, int out_size, void* d_ws, size_t ws_size,
                              hipStream_t stream) {
    const float* nh  = (const float*)d_in[0];   // (16384, 2, 128)
    const float* A   = (const float*)d_in[1];   // (128, 128, 384)
    const float* U1w = (const float*)d_in[2];   // (384, 128)
    const float* U2w = (const float*)d_in[3];   // (384, 128)
    const float* U2b = (const float*)d_in[4];   // (384,)
    float* out = (float*)d_out;                 // (16384, 384)

    const size_t need = (size_t)NFRAG_TOT * 1024;   // ~12.2 MB of bf16 fragments
    if (ws_size >= need) {
        unsigned short* ws = (unsigned short*)d_ws;
        btc_prep_kernel<<<(NFRAG_TOT + 7) / 8, 256, 0, stream>>>(A, U1w, U2w, ws);
        dim3 grid(NNODES / BM, NOUT / 128);                    // 128 x 3
        const size_t lds = (size_t)2 * BM * LDH * sizeof(float);  // 132 KB
        btc_main_kernel<<<grid, 256, lds, stream>>>(nh, ws, U2b, out);
    } else {
        btc_fallback_kernel<<<NNODES, 128, 0, stream>>>(nh, A, U1w, U2w, U2b, out);
    }
}
